// Local_Receptive_Field_18872086298823
// MI455X (gfx1250) — matrix operand order repaired and checked by reading, not compiled, not run on hardware
//
#include <hip/hip_runtime.h>
#include <hip/hip_bf16.h>

typedef __attribute__((ext_vector_type(16))) _Float16 v16h;
typedef __attribute__((ext_vector_type(8)))  _Float16 v8h;
typedef __attribute__((ext_vector_type(4)))  _Float16 v4h;
typedef __attribute__((ext_vector_type(8)))  float    v8f;

#define NB   1024   // batch
#define NN   1024   // nodes
#define HIDP 96     // 90 padded to 96
#define HID  90

static __device__ inline v16h cat8(v8h lo, v8h hi) {
    v16h r;
#pragma unroll
    for (int i = 0; i < 8; ++i) { r[i] = lo[i]; r[8 + i] = hi[i]; }
    return r;
}

static __device__ inline float elu1(float x) {
    return x > 0.f ? x : (__expf(x) - 1.f);
}

// ---------------- prep kernels ----------------

// cb[n][c] = b1[c] + sum_j controls[n][j] * W1[control_list[n][j]][c]   (c<90, else 0)
__global__ void prep_cb(const float* __restrict__ controls,
                        const float* __restrict__ W1,
                        const float* __restrict__ b1,
                        const int*   __restrict__ clist,
                        float* __restrict__ cb) {
    int idx = blockIdx.x * blockDim.x + threadIdx.x;
    if (idx >= NN * HIDP) return;
    int n = idx / HIDP, c = idx % HIDP;
    float v = 0.f;
    if (c < HID) {
        v = b1[c];
#pragma unroll
        for (int j = 0; j < 3; ++j)
            v += controls[n * 3 + j] * W1[clist[n * 3 + j] * HID + c];
    }
    cb[idx] = v;
}

// wd[n][c][i] = W1[inputs_list[n][i]][c]  (f16, c<90 else 0)
__global__ void prep_wd(const float* __restrict__ W1,
                        const int*   __restrict__ ilist,
                        _Float16* __restrict__ wd) {
    int idx = blockIdx.x * blockDim.x + threadIdx.x;
    if (idx >= NN * HIDP) return;
    int n = idx / HIDP, c = idx % HIDP;
#pragma unroll
    for (int i = 0; i < 4; ++i) {
        float v = (c < HID) ? W1[ilist[n * 4 + i] * HID + c] : 0.f;
        wd[idx * 4 + i] = (_Float16)v;
    }
}

// w2t[c][k] = W2[k][c] (f16, padded); b2p/w3p padded f32
__global__ void prep_w2(const float* __restrict__ W2,
                        const float* __restrict__ b2,
                        const float* __restrict__ W3,
                        _Float16* __restrict__ w2t,
                        float* __restrict__ b2p,
                        float* __restrict__ w3p) {
    int idx = blockIdx.x * blockDim.x + threadIdx.x;
    if (idx >= HIDP * HIDP) return;
    int c = idx / HIDP, k = idx % HIDP;
    w2t[idx] = (_Float16)((c < HID && k < HID) ? W2[k * HID + c] : 0.f);
    if (idx < HIDP) {
        b2p[idx] = (idx < HID) ? b2[idx] : 0.f;
        w3p[idx] = (idx < HID) ? W3[idx] : 0.f;
    }
}

// ---------------- main kernel ----------------
// One wave per (node n, 16-batch tile). 8 waves / 256-thread block.
__global__ __launch_bounds__(256) void dnpu_mlp(
    const float*    __restrict__ x,    // [B,1,64,64]
    const float*    __restrict__ cb,   // [N,96]
    const _Float16* __restrict__ wd,   // [N,96,4]
    const _Float16* __restrict__ w2t,  // [96,96]
    const float*    __restrict__ b2p,  // [96]
    const float*    __restrict__ w3p,  // [96]
    const float*    __restrict__ b3,   // [1]
    float*          __restrict__ out)  // [B,N]
{
    __shared__ __attribute__((aligned(16))) _Float16 h1s[8][16 * HIDP];

    const int wave = threadIdx.x >> 5;
    const int lane = threadIdx.x & 31;
    const int gw   = blockIdx.x * 8 + wave;       // 0 .. 65535
    const int n    = gw & (NN - 1);               // node
    const int b0   = (gw >> 10) << 4;             // batch tile base
    const int half = lane >> 4;                   // 0: lanes 0-15, 1: lanes 16-31
    const int lid  = lane & 15;

    // ---- layer-1 A fragment: 16x32 f16, K = patch[0..3], rest zero ----
    v16h a1;
#pragma unroll
    for (int i = 0; i < 16; ++i) a1[i] = (_Float16)0.f;
    if (half == 0) {
        const int r = n >> 5, c0 = n & 31;        // node = r*32 + c0
        const float* px = x + (((size_t)(b0 + lid) * 64 + 2 * r) * 64 + 2 * c0);
        float2 p0 = *(const float2*)px;           // kh=0: kw=0,1
        float2 p1 = *(const float2*)(px + 64);    // kh=1: kw=0,1
        a1[0] = (_Float16)p0.x; a1[1] = (_Float16)p0.y;
        a1[2] = (_Float16)p1.x; a1[3] = (_Float16)p1.y;
    }

    _Float16* myh1 = h1s[wave];

    // ---- layer 1: 6 WMMAs (K=32, only 4 live), ELU, transpose via LDS ----
#pragma unroll
    for (int nt = 0; nt < 6; ++nt) {
        const int col = nt * 16 + lid;
        v16h bf;
#pragma unroll
        for (int i = 0; i < 16; ++i) bf[i] = (_Float16)0.f;
        if (half == 0) {                          // lanes 0-15 hold the live K=0..3
            v4h w4 = *(const v4h*)(wd + ((size_t)n * HIDP + col) * 4);
#pragma unroll
            for (int i = 0; i < 4; ++i) bf[i] = w4[i];
        }
        const float cbv = cb[n * HIDP + col];
        v8f acc;
#pragma unroll
        for (int v = 0; v < 8; ++v) acc[v] = cbv;
        acc = __builtin_amdgcn_wmma_f32_16x16x32_f16(
            false, a1, false, bf, (short)0, acc, false, false);
        // D layout: lane holds rows {half*8 + v}, column `col`
        const int row0 = half * 8;
#pragma unroll
        for (int v = 0; v < 8; ++v)
            myh1[(row0 + v) * HIDP + col] = (_Float16)elu1(acc[v]);
    }
    asm volatile("s_wait_dscnt 0" ::: "memory");

    // ---- reload h1 as three A fragments (row = lid, K packed per ISA layout) ----
    v16h A2[3];
#pragma unroll
    for (int kc = 0; kc < 3; ++kc) {
        const _Float16* hp = myh1 + lid * HIDP + kc * 32 + half * 8;
        v8h lo = *(const v8h*)hp;          // K = kc*32 + kb .. +7
        v8h hi = *(const v8h*)(hp + 16);   // K = kc*32 + 16 + kb .. +7
        A2[kc] = cat8(lo, hi);
    }

    // ---- layer 2 (18 WMMAs) fused with layer-3 dot ----
    float p[8];
#pragma unroll
    for (int v = 0; v < 8; ++v) p[v] = 0.f;

#pragma unroll
    for (int nt = 0; nt < 6; ++nt) {
        const int col = nt * 16 + lid;
        const float bb = b2p[col];
        v8f acc;
#pragma unroll
        for (int v = 0; v < 8; ++v) acc[v] = bb;
#pragma unroll
        for (int kc = 0; kc < 3; ++kc) {
            // B fragment: lane = column, K = kb..kb+7 and 16+kb..16+kb+7 per lane-half (kb = half*8)
            const _Float16* bp = w2t + (size_t)col * HIDP + kc * 32 + half * 8;
            v8h lo = *(const v8h*)bp;
            v8h hi = *(const v8h*)(bp + 16);
            acc = __builtin_amdgcn_wmma_f32_16x16x32_f16(
                false, A2[kc], false, cat8(lo, hi), (short)0, acc, false, false);
        }
        const float w3v = w3p[col];
#pragma unroll
        for (int v = 0; v < 8; ++v)
            p[v] += elu1(acc[v]) * w3v;
    }

    // ---- reduce columns across each 16-lane half ----
#pragma unroll
    for (int m = 1; m < 16; m <<= 1)
#pragma unroll
        for (int v = 0; v < 8; ++v)
            p[v] += __shfl_xor(p[v], m, 32);

    if (lid == 0) {
        const float bias = b3[0];
        const int row0 = half * 8;
#pragma unroll
        for (int v = 0; v < 8; ++v)
            out[(size_t)(b0 + row0 + v) * NN + n] = p[v] + bias;
    }
}

// ---------------- launcher ----------------
extern "C" void kernel_launch(void* const* d_in, const int* in_sizes, int n_in,
                              void* d_out, int out_size, void* d_ws, size_t ws_size,
                              hipStream_t stream) {
    const float* x        = (const float*)d_in[0];
    const float* controls = (const float*)d_in[1];
    const float* W1       = (const float*)d_in[2];
    const float* b1       = (const float*)d_in[3];
    const float* W2       = (const float*)d_in[4];
    const float* b2       = (const float*)d_in[5];
    const float* W3       = (const float*)d_in[6];
    const float* b3       = (const float*)d_in[7];
    const int*   ilist    = (const int*)d_in[8];
    const int*   clist    = (const int*)d_in[9];
    float* out = (float*)d_out;

    // workspace layout (all offsets 16B-aligned)
    char* w = (char*)d_ws;
    float*    cb  = (float*)(w + 0);                      // 1024*96*4  = 393216
    _Float16* wd  = (_Float16*)(w + 393216);              // 1024*96*4*2= 786432
    _Float16* w2t = (_Float16*)(w + 393216 + 786432);     // 96*96*2    = 18432
    float*    b2p = (float*)(w + 393216 + 786432 + 18432);        // 384
    float*    w3p = (float*)(w + 393216 + 786432 + 18432 + 384);  // 384
    (void)ws_size; (void)in_sizes; (void)n_in; (void)out_size;

    prep_cb<<<(NN * HIDP + 255) / 256, 256, 0, stream>>>(controls, W1, b1, clist, cb);
    prep_wd<<<(NN * HIDP + 255) / 256, 256, 0, stream>>>(W1, ilist, wd);
    prep_w2<<<(HIDP * HIDP + 255) / 256, 256, 0, stream>>>(W2, b2, W3, w2t, b2p, w3p);

    // 65536 wave-tiles, 8 waves per 256-thread block
    dnpu_mlp<<<65536 / 8, 256, 0, stream>>>(x, cb, wd, w2t, b2p, w3p, b3, out);
}
